// RobustGCN_82197084110911
// MI455X (gfx1250) — matrix operand order repaired and checked by reading, not compiled, not run on hardware
//
#include <hip/hip_runtime.h>
#include <hip/hip_bf16.h>

// ---------------------------------------------------------------------------
// RobustGCN forward for MI455X (gfx1250, wave32).
//  - GEMMs: v_wmma_f32_16x16x32_bf16 with bf16 hi/lo split (fp32-class
//    accuracy). Weights pre-packed into WMMA B-fragment layout (bf16 hi/lo)
//    so the inner loop is pure b128 loads + WMMA. Activations converted once
//    per row-tile and reused across all column tiles (register blocking).
//  - Scatter: coalesced global_atomic_add_f32 stream (bandwidth-dominant).
//  - Output eps reproduces JAX threefry2x32(key=42) + erfinv normal sampling.
// ---------------------------------------------------------------------------

typedef __attribute__((ext_vector_type(16))) __bf16 v16bf;
typedef __attribute__((ext_vector_type(8)))  float  v8f;

static __device__ __forceinline__ void bf16_split(float f, v16bf& hi, v16bf& lo, int e) {
  __bf16 h = (__bf16)f;
  hi[e] = h;
  lo[e] = (__bf16)(f - (float)h);
}

// A fragment: 16x32 (MxK) bf16. Lane L holds row M = L%16; K-half kh = L/16.
// elements 0..7  -> K = k0 + kh*16 + e       (contiguous)
// elements 8..15 -> K = k0 + kh*16 + e       (contiguous)
static __device__ __forceinline__ void load_a_frag(const float* __restrict__ arow,
                                                   int k0, int kh,
                                                   v16bf& hi, v16bf& lo) {
  const float* p0 = arow + k0 + kh * 16;
  const float* p1 = p0 + 8;
#pragma unroll
  for (int e = 0; e < 8; ++e) bf16_split(p0[e], hi, lo, e);
#pragma unroll
  for (int e = 0; e < 8; ++e) bf16_split(p1[e], hi, lo, e + 8);
}

// acc += A*B with bf16 hi/lo compensation (drop lo*lo term).
static __device__ __forceinline__ v8f wmma3(v8f acc, v16bf ah, v16bf al,
                                            v16bf bh, v16bf bl) {
  acc = __builtin_amdgcn_wmma_f32_16x16x32_bf16(false, ah, false, bh, (short)0, acc, false, false);
  acc = __builtin_amdgcn_wmma_f32_16x16x32_bf16(false, ah, false, bl, (short)0, acc, false, false);
  acc = __builtin_amdgcn_wmma_f32_16x16x32_bf16(false, al, false, bh, (short)0, acc, false, false);
  return acc;
}

// ---------------------------------------------------------------------------
// Pre-pack f32 weights (K x N row-major) into WMMA B-fragment layout, bf16
// hi/lo:  P[((kb*N + n)*2 + kh)*16 + e] = W[(kb*32 + kh*16 + e)*N + n].
// Lane (kh, n) of a wave then loads its 16-element fragment as one contiguous
// 32-byte chunk (32B aligned).
// ---------------------------------------------------------------------------
__global__ void __launch_bounds__(256)
rgcn_pack_w(const float* __restrict__ W, __bf16* __restrict__ PH,
            __bf16* __restrict__ PL, int K, int N)
{
  const int total = K * N;
  const int i = blockIdx.x * blockDim.x + threadIdx.x;
  if (i >= total) return;
  const int e  = i & 15;
  int t        = i >> 4;
  const int kh = t & 1;
  t >>= 1;
  const int n  = t % N;
  const int kb = t / N;
  const int k  = kb * 32 + kh * 16 + e;
  const float f = W[(size_t)k * N + n];
  const __bf16 h = (__bf16)f;
  PH[i] = h;
  PL[i] = (__bf16)(f - (float)h);
}

// ---------------------------------------------------------------------------
// Fused dual-GEMM + attention epilogue. One wave per 16-row tile; all CT
// column tiles register-blocked so each A fragment feeds 2*CT WMMA-triples.
// SAME=true: inM==inV (layer 1, input x shared by mean and var paths).
// ---------------------------------------------------------------------------
template<int CT, bool SAME>
__global__ void __launch_bounds__(256)
rgcn_gemm_msgs(const float* __restrict__ inM, const float* __restrict__ inV,
               const __bf16* __restrict__ pMh, const __bf16* __restrict__ pMl,
               const __bf16* __restrict__ pVh, const __bf16* __restrict__ pVl,
               const float* __restrict__ bM,  const float* __restrict__ bV,
               const float* __restrict__ norm1, const float* __restrict__ norm2,
               float* __restrict__ outM, float* __restrict__ outV,
               int K, int rowTiles, float gamma)
{
  const int N = CT * 16;
  const int wave = blockIdx.x * (blockDim.x >> 5) + (threadIdx.x >> 5);
  if (wave >= rowTiles) return;                    // wave-uniform: EXEC stays full

  const int lane = threadIdx.x & 31;
  const int kh   = lane >> 4;
  const int nloc = lane & 15;
  const int mrow = wave * 16 + nloc;               // A-matrix row for this lane

  const float* am = inM + (size_t)mrow * K;
  const float* av = inV + (size_t)mrow * K;

  v8f accM[CT], accV[CT];
#pragma unroll
  for (int c = 0; c < CT; ++c) { accM[c] = (v8f){}; accV[c] = (v8f){}; }

  for (int k0 = 0, kb = 0; k0 < K; k0 += 32, ++kb) {
    v16bf amh, aml, avh, avl;
    load_a_frag(am, k0, kh, amh, aml);
    if (!SAME) load_a_frag(av, k0, kh, avh, avl);

#pragma unroll
    for (int c = 0; c < CT; ++c) {
      const int n = c * 16 + nloc;
      const size_t off = (((size_t)kb * N + n) * 2 + kh) * 16;   // 32B aligned
      const v16bf bmh = *(const v16bf*)(pMh + off);
      const v16bf bml = *(const v16bf*)(pMl + off);
      const v16bf bvh = *(const v16bf*)(pVh + off);
      const v16bf bvl = *(const v16bf*)(pVl + off);
      accM[c] = wmma3(accM[c], amh, aml, bmh, bml);
      if (SAME) accV[c] = wmma3(accV[c], amh, aml, bvh, bvl);
      else      accV[c] = wmma3(accV[c], avh, avl, bvh, bvl);
    }
  }

  // Epilogue: relu -> att = exp(-gamma*var) -> pre-scaled messages.
#pragma unroll
  for (int r = 0; r < 8; ++r) {
    const int row = wave * 16 + kh * 8 + r;        // C/D row layout (ISA 7.12.2)
    const float n1 = norm1[row];
    const float n2 = norm2[row];
#pragma unroll
    for (int c = 0; c < CT; ++c) {
      const int n = c * 16 + nloc;
      float mean = accM[c][r] + bM[n];  mean = mean > 0.f ? mean : 0.f;
      float var  = accV[c][r] + bV[n];  var  = var  > 0.f ? var  : 0.f;
      const float att = expf(-gamma * var);
      outM[(size_t)row * N + n] = mean * att * n1;
      outV[(size_t)row * N + n] = var  * att * att * n2;
    }
  }
}

// ---------------------------------------------------------------------------
// Bandwidth-dominant phase: gather at src, scatter-add at dst.
// Threads within one edge's feature span share src/dst (broadcast loads) and
// touch contiguous features (coalesced loads + global_atomic_add_f32).
// ---------------------------------------------------------------------------
__global__ void __launch_bounds__(256)
rgcn_scatter(const int* __restrict__ src, const int* __restrict__ dst,
             const float* __restrict__ m, const float* __restrict__ v,
             float* __restrict__ accM, float* __restrict__ accV,
             int nEdges, int logF)
{
  const int F = 1 << logF;
  const long long total  = (long long)nEdges << logF;
  const long long stride = (long long)gridDim.x * blockDim.x;
  for (long long i = (long long)blockIdx.x * blockDim.x + threadIdx.x;
       i < total; i += stride) {
    const int e = (int)(i >> logF);
    const int f = (int)i & (F - 1);
    const int s = src[e];
    const int d = dst[e];
    atomicAdd(accM + (size_t)d * F + f, m[(size_t)s * F + f]);
    atomicAdd(accV + (size_t)d * F + f, v[(size_t)s * F + f]);
  }
}

__global__ void __launch_bounds__(256)
rgcn_zero(float* __restrict__ p, long long count)
{
  const long long stride = (long long)gridDim.x * blockDim.x;
  for (long long i = (long long)blockIdx.x * blockDim.x + threadIdx.x;
       i < count; i += stride)
    p[i] = 0.0f;
}

__global__ void __launch_bounds__(256)
rgcn_degrees(const int* __restrict__ dst, float* __restrict__ deg, int nEdges)
{
  const long long stride = (long long)gridDim.x * blockDim.x;
  for (long long e = (long long)blockIdx.x * blockDim.x + threadIdx.x;
       e < nEdges; e += stride)
    atomicAdd(deg + dst[e], 1.0f);
}

// deg (stored in norm1) -> norm1 = deg^-1/2, norm2 = deg^-1  (deg clipped >=1)
__global__ void __launch_bounds__(256)
rgcn_norms(float* __restrict__ norm1, float* __restrict__ norm2, int nNodes)
{
  const long long stride = (long long)gridDim.x * blockDim.x;
  for (long long i = (long long)blockIdx.x * blockDim.x + threadIdx.x;
       i < nNodes; i += stride) {
    const float d = fmaxf(norm1[i], 1.0f);
    const float r = rsqrtf(d);
    norm1[i] = r;
    norm2[i] = r * r;
  }
}

__global__ void __launch_bounds__(256)
rgcn_finalize(const float* __restrict__ accM, const float* __restrict__ accV,
              const float* __restrict__ norm1, const float* __restrict__ norm2,
              float* __restrict__ outM, float* __restrict__ outV,
              int nNodes, int logF, int doRelu)
{
  const long long total  = (long long)nNodes << logF;
  const long long stride = (long long)gridDim.x * blockDim.x;
  for (long long i = (long long)blockIdx.x * blockDim.x + threadIdx.x;
       i < total; i += stride) {
    const int node = (int)(i >> logF);
    float mv = accM[i] * norm1[node];
    float vv = accV[i] * norm2[node];
    if (doRelu) { mv = fmaxf(mv, 0.f); vv = fmaxf(vv, 0.f); }
    outM[i] = mv;
    outV[i] = vv;
  }
}

// --------------------------- JAX threefry2x32 ------------------------------
static __device__ __forceinline__ unsigned rotl32(unsigned x, int n) {
  return (x << n) | (x >> (32 - n));
}

static __device__ void threefry2x32(unsigned k0, unsigned k1,
                                    unsigned x0, unsigned x1,
                                    unsigned* o0, unsigned* o1)
{
  const unsigned k2 = k0 ^ k1 ^ 0x1BD11BDAu;
  const int ra[4] = {13, 15, 26, 6};
  const int rb[4] = {17, 29, 16, 24};
  x0 += k0; x1 += k1;
#pragma unroll
  for (int i = 0; i < 4; ++i) { x0 += x1; x1 = rotl32(x1, ra[i]); x1 ^= x0; }
  x0 += k1; x1 += k2 + 1u;
#pragma unroll
  for (int i = 0; i < 4; ++i) { x0 += x1; x1 = rotl32(x1, rb[i]); x1 ^= x0; }
  x0 += k2; x1 += k0 + 2u;
#pragma unroll
  for (int i = 0; i < 4; ++i) { x0 += x1; x1 = rotl32(x1, ra[i]); x1 ^= x0; }
  x0 += k0; x1 += k1 + 3u;
#pragma unroll
  for (int i = 0; i < 4; ++i) { x0 += x1; x1 = rotl32(x1, rb[i]); x1 ^= x0; }
  x0 += k1; x1 += k2 + 4u;
#pragma unroll
  for (int i = 0; i < 4; ++i) { x0 += x1; x1 = rotl32(x1, ra[i]); x1 ^= x0; }
  x0 += k2; x1 += k0 + 5u;
  *o0 = x0; *o1 = x1;
}

// Layer-2 finalize fused with eps*sqrt(var+1e-8)+mean output.
// eps mimics jax.random.normal(key(42)): threefry bits over split iota
// counters -> uniform in (-1,1) -> sqrt(2)*erfinv.
__global__ void __launch_bounds__(256)
rgcn_output(const float* __restrict__ accM, const float* __restrict__ accV,
            const float* __restrict__ norm1, const float* __restrict__ norm2,
            float* __restrict__ out, int nNodes, int logF)
{
  const long long total  = (long long)nNodes << logF;
  const unsigned  half   = (unsigned)(total >> 1);
  const long long stride = (long long)gridDim.x * blockDim.x;
  for (long long i = (long long)blockIdx.x * blockDim.x + threadIdx.x;
       i < total; i += stride) {
    const int node = (int)(i >> logF);
    const float mean = accM[i] * norm1[node];
    const float var  = accV[i] * norm2[node];

    const unsigned li = (unsigned)i;
    unsigned x0, x1; int pick;
    if (li < half) { x0 = li;        x1 = li + half; pick = 0; }
    else           { x0 = li - half; x1 = li;        pick = 1; }
    unsigned o0, o1;
    threefry2x32(0u, 42u, x0, x1, &o0, &o1);
    const unsigned bits = pick ? o1 : o0;

    const float u01 = __uint_as_float((bits >> 9) | 0x3F800000u) - 1.0f; // [0,1)
    const float mn  = -0.99999994f;                                      // nextafter(-1,0)
    const float u   = u01 * (1.0f - mn) + mn;                            // (-1,1)
    const float eps = 1.4142135623730951f * erfinvf(u);

    out[i] = eps * sqrtf(var + 1e-8f) + mean;
  }
}

// ---------------------------------------------------------------------------
static inline int cdiv(long long a, long long b) { return (int)((a + b - 1) / b); }

extern "C" void kernel_launch(void* const* d_in, const int* in_sizes, int n_in,
                              void* d_out, int out_size, void* d_ws, size_t ws_size,
                              hipStream_t stream)
{
  const float* x   = (const float*)d_in[0];
  const int*   src = (const int*)  d_in[1];
  const int*   dst = (const int*)  d_in[2];
  const float* wm1 = (const float*)d_in[3];
  const float* wv1 = (const float*)d_in[4];
  const float* bm1 = (const float*)d_in[5];
  const float* bv1 = (const float*)d_in[6];
  const float* wm2 = (const float*)d_in[7];
  const float* wv2 = (const float*)d_in[8];
  const float* bm2 = (const float*)d_in[9];
  const float* bv2 = (const float*)d_in[10];

  const int HID    = in_sizes[5];              // 128
  const int OUTF   = in_sizes[9];              // 64
  const int IN_F   = in_sizes[3] / HID;        // 256
  const int nNodes = in_sizes[0] / IN_F;       // 100000
  const int nEdges = in_sizes[1];              // 1600000
  const float gamma = 1.0f;

  // Workspace layout (floats): norm1|norm2|A|B|C|D | packed bf16 weights.
  float* ws    = (float*)d_ws;
  float* norm1 = ws;
  float* norm2 = norm1 + nNodes;
  float* bufA  = norm2 + nNodes;
  float* bufB  = bufA + (size_t)nNodes * HID;
  float* bufC  = bufB + (size_t)nNodes * HID;
  float* bufD  = bufC + (size_t)nNodes * HID;

  const size_t w1n = (size_t)IN_F * HID;       // 32768
  const size_t w2n = (size_t)HID * OUTF;       // 8192
  __bf16* pw    = (__bf16*)(bufD + (size_t)nNodes * HID);
  __bf16* pM1h  = pw;             __bf16* pM1l = pM1h + w1n;
  __bf16* pV1h  = pM1l + w1n;     __bf16* pV1l = pV1h + w1n;
  __bf16* pM2h  = pV1l + w1n;     __bf16* pM2l = pM2h + w2n;
  __bf16* pV2h  = pM2l + w2n;     __bf16* pV2l = pV2h + w2n;

  const int T = 256;
  const int gridBig = 4096;

  // ---- pack weights into WMMA fragment layout (tiny, L2-resident) ----
  rgcn_pack_w<<<cdiv(w1n, T), T, 0, stream>>>(wm1, pM1h, pM1l, IN_F, HID);
  rgcn_pack_w<<<cdiv(w1n, T), T, 0, stream>>>(wv1, pV1h, pV1l, IN_F, HID);
  rgcn_pack_w<<<cdiv(w2n, T), T, 0, stream>>>(wm2, pM2h, pM2l, HID, OUTF);
  rgcn_pack_w<<<cdiv(w2n, T), T, 0, stream>>>(wv2, pV2h, pV2l, HID, OUTF);

  // ---- degrees -> norms ----
  rgcn_zero   <<<cdiv(nNodes, T), T, 0, stream>>>(norm1, (long long)nNodes);
  rgcn_degrees<<<cdiv(nEdges, T), T, 0, stream>>>(dst, norm1, nEdges);
  rgcn_norms  <<<cdiv(nNodes, T), T, 0, stream>>>(norm1, norm2, nNodes);

  const int rowTiles = nNodes / 16;

  // ---- layer 1: dual GEMM (K=IN_F, N=HID=8 tiles, shared input x) ----
  rgcn_gemm_msgs<8, true><<<cdiv(rowTiles, T / 32), T, 0, stream>>>(
      x, x, pM1h, pM1l, pV1h, pV1l, bm1, bv1,
      norm1, norm2, bufA, bufB, IN_F, rowTiles, gamma);

  // ---- scatter-add into C,D ----
  rgcn_zero<<<gridBig, T, 0, stream>>>(bufC, 2LL * nNodes * HID);  // C,D contiguous
  rgcn_scatter<<<gridBig, T, 0, stream>>>(src, dst, bufA, bufB, bufC, bufD, nEdges, 7);
  // ---- finalize layer 1 (relu) -> A,B become layer-2 inputs ----
  rgcn_finalize<<<gridBig, T, 0, stream>>>(bufC, bufD, norm1, norm2, bufA, bufB,
                                           nNodes, 7, 1);

  // ---- layer 2: dual GEMM (K=HID, N=OUTF=4 tiles, distinct inputs) ----
  float* m2    = bufC;
  float* v2    = bufC + (size_t)nNodes * OUTF;
  float* accM2 = bufD;
  float* accV2 = bufD + (size_t)nNodes * OUTF;
  rgcn_gemm_msgs<4, false><<<cdiv(rowTiles, T / 32), T, 0, stream>>>(
      bufA, bufB, pM2h, pM2l, pV2h, pV2l, bm2, bv2,
      norm1, norm2, m2, v2, HID, rowTiles, gamma);

  rgcn_zero<<<gridBig, T, 0, stream>>>(bufD, 2LL * nNodes * OUTF);
  rgcn_scatter<<<gridBig, T, 0, stream>>>(src, dst, m2, v2, accM2, accV2, nEdges, 6);

  // ---- finalize layer 2 (no relu) fused with reparameterization output ----
  rgcn_output<<<gridBig, T, 0, stream>>>(accM2, accV2, norm1, norm2,
                                         (float*)d_out, nNodes, 6);
}
